// TabNet_90898687853206
// MI455X (gfx1250) — matrix operand order repaired and checked by reading, not compiled, not run on hardware
//
#include <hip/hip_runtime.h>
#include <hip/hip_bf16.h>

// ---------------------------------------------------------------------------
// TabNet fused forward for MI455X (gfx1250), round 2.
//
// Roofline: ~137 GFLOP, ~140 MB HBM traffic (6 us @ 23.3 TB/s) -> the real
// limiter is L2->WGP weight-fragment bandwidth. M=32 rows per wave doubles
// FLOP per L2 byte vs round 1 (each B fragment feeds two 16x16x32 WMMAs).
// All activations LDS-resident (112 KB dynamic LDS, 2 waves/WGP of 320 KB).
// Weights pre-converted once to bf16 N-major in d_ws (L2-resident, 0.6 MB).
// ---------------------------------------------------------------------------

typedef __attribute__((ext_vector_type(16))) __bf16 v16bf;
typedef __attribute__((ext_vector_type(8)))  float  v8f;

#define TB_B 131072
#define TB_D 256
#define TB_H 128
#define TB_C 10
#define TB_S 3
#define TB_GAMMA 1.5f
#define TB_EPS 1e-5f
#define ROWS 32                        // rows per wave (2 WMMA M-tiles)

// bf16 offsets inside d_ws (element counts)
#define OFF_SH 0                        // sh_wT  [256][256]
#define OFF_FT (256 * 256)              // ft_wT  [4][256][128]
#define OFF_AT (OFF_FT + 4 * 256 * 128) // at_wT  [3][256][128]
#define TOT_W  (OFF_AT + 3 * 256 * 128)

// dynamic LDS layout (bytes)
#define L_XS    0                       // bf16 [32][256]  16 KB
#define L_ABUF  (16 * 1024)             // bf16 [32][256]  16 KB
#define L_YBUF  (32 * 1024)             // f32  [32][256]  32 KB
#define L_PRIOR (64 * 1024)             // f32  [32][256]  32 KB
#define L_OAGG  (96 * 1024)             // f32  [32][128]  16 KB
#define L_TOTAL (112 * 1024)

__device__ __forceinline__ unsigned short f2bf(float f) {
    unsigned u = __builtin_bit_cast(unsigned, f);
    u += 0x7FFFu + ((u >> 16) & 1u);   // round-to-nearest-even
    return (unsigned short)(u >> 16);
}
__device__ __forceinline__ float bf2f(unsigned short h) {
    unsigned u = ((unsigned)h) << 16;
    return __builtin_bit_cast(float, u);
}

// --------------------------- weight conversion -----------------------------
// f32 row-major [K][N] -> bf16 N-major [N][K]: each WMMA B-fragment becomes
// two contiguous 16B global loads per lane.
__global__ __launch_bounds__(256) void tabnet_convert_wts(
    const float* __restrict__ sh_w,   // [256][256]
    const float* __restrict__ ft_w,   // [4][128][256]
    const float* __restrict__ at_w,   // [3][128][256]
    unsigned short* __restrict__ out)
{
    int t = blockIdx.x * blockDim.x + threadIdx.x;
    if (t < OFF_FT) {
        int n = t >> 8, k = t & 255;
        out[t] = f2bf(sh_w[k * 256 + n]);
    } else if (t < OFF_AT) {
        int u = t - OFF_FT;
        int i = u >> 15, rem = u & 32767;
        int n = rem >> 7, k = rem & 127;
        out[t] = f2bf(ft_w[i * 32768 + k * 256 + n]);
    } else if (t < TOT_W) {
        int u = t - OFF_AT;
        int s = u >> 15, rem = u & 32767;
        int n = rem >> 7, k = rem & 127;
        out[t] = f2bf(at_w[s * 32768 + k * 256 + n]);
    }
}

// ------------------------------ fragments ----------------------------------
// A (16x32 bf16) from LDS, documented layout:
//  lanes 0-15 : M=lane,    K = k0+{0..7, 16..23}
//  lanes 16-31: M=lane-16, K = k0+{8..15,24..31}
__device__ __forceinline__ v16bf load_a(const unsigned short (*A)[256], int lane, int k0) {
    int m = lane & 15, half = lane >> 4;
    union { uint4 u[2]; v16bf v; } t;
    t.u[0] = *(const uint4*)&A[m][k0 + half * 8];
    t.u[1] = *(const uint4*)&A[m][k0 + 16 + half * 8];
    return t.v;
}
// B (32x16 bf16) from N-major bf16 weights:
//  lane l: N = n0 + (l&15); lanes 0-15 K=k0+{0..7,16..23}, lanes 16-31 K=k0+{8..15,24..31}
__device__ __forceinline__ v16bf load_b(const unsigned short* __restrict__ WT,
                                        int K, int lane, int n0, int k0) {
    int n = n0 + (lane & 15), half = lane >> 4;
    const unsigned short* p = WT + (size_t)n * K + k0 + half * 8;
    union { uint4 u[2]; v16bf v; } t;
    t.u[0] = *(const uint4*)p;
    t.u[1] = *(const uint4*)(p + 16);
    return t.v;
}

// Y[32][N] = A[32][K] @ W[K][N] + bias. Two M-tiles share every B fragment
// (2x FLOP per L2 byte). C layout: VGPR r -> M = r + 8*(lane>=16),
// N = n0 + (lane&15).
__device__ __forceinline__ void wmma_gemm(const unsigned short (*A)[256],
                                          const unsigned short* __restrict__ WT,
                                          int K, int N,
                                          const float* __restrict__ bias,
                                          float (*Y)[256], int lane)
{
    for (int n0 = 0; n0 < N; n0 += 16) {
        v8f acc0 = {};
        v8f acc1 = {};
        for (int k0 = 0; k0 < K; k0 += 32) {
            v16bf b  = load_b(WT, K, lane, n0, k0);
            v16bf a0 = load_a(A,      lane, k0);
            v16bf a1 = load_a(A + 16, lane, k0);
            acc0 = __builtin_amdgcn_wmma_f32_16x16x32_bf16(
                false, a0, false, b, (short)0, acc0, false, false);
            acc1 = __builtin_amdgcn_wmma_f32_16x16x32_bf16(
                false, a1, false, b, (short)0, acc1, false, false);
        }
        int n = n0 + (lane & 15);
        float bv = bias[n];
        int mbase = (lane >> 4) * 8;
#pragma unroll
        for (int r = 0; r < 8; ++r) {
            Y[mbase + r][n]      = acc0[r] + bv;
            Y[16 + mbase + r][n] = acc1[r] + bv;
        }
    }
}

// LayerNorm over 256 cols + GLU: h = a * sigmoid(g). Writes bf16 h into
// Aout[.][0..127]; optionally accumulates into agg.
__device__ __forceinline__ void ln_glu(float (*Y)[256],
                                       const float* __restrict__ lw,
                                       const float* __restrict__ lb,
                                       unsigned short (*Aout)[256],
                                       float (*agg)[128],
                                       int lane, bool accumulate)
{
    for (int r = 0; r < ROWS; ++r) {
        float s = 0.f, s2 = 0.f;
#pragma unroll
        for (int i = 0; i < 8; ++i) {
            float v = Y[r][lane + 32 * i];
            s += v; s2 += v * v;
        }
#pragma unroll
        for (int o = 16; o; o >>= 1) { s += __shfl_xor(s, o); s2 += __shfl_xor(s2, o); }
        float m   = s * (1.f / 256.f);
        float var = s2 * (1.f / 256.f) - m * m;
        float rs  = rsqrtf(var + TB_EPS);
#pragma unroll
        for (int i = 0; i < 4; ++i) {
            int j = lane + 32 * i, j2 = j + 128;
            float a = (Y[r][j]  - m) * rs * lw[j]  + lb[j];
            float g = (Y[r][j2] - m) * rs * lw[j2] + lb[j2];
            float h = a * (1.f / (1.f + __expf(-g)));
            Aout[r][j] = f2bf(h);
            if (accumulate) agg[r][j] += h;
        }
    }
}

// Plain LayerNorm over 256 cols, in place (attentive transformer).
__device__ __forceinline__ void ln_inplace(float (*Y)[256],
                                           const float* __restrict__ lw,
                                           const float* __restrict__ lb, int lane)
{
    for (int r = 0; r < ROWS; ++r) {
        float s = 0.f, s2 = 0.f;
#pragma unroll
        for (int i = 0; i < 8; ++i) {
            float v = Y[r][lane + 32 * i];
            s += v; s2 += v * v;
        }
#pragma unroll
        for (int o = 16; o; o >>= 1) { s += __shfl_xor(s, o); s2 += __shfl_xor(s2, o); }
        float m = s * (1.f / 256.f);
        float var = s2 * (1.f / 256.f) - m * m;
        float rs = rsqrtf(var + TB_EPS);
#pragma unroll
        for (int i = 0; i < 8; ++i) {
            int j = lane + 32 * i;
            Y[r][j] = (Y[r][j] - m) * rs * lw[j] + lb[j];
        }
    }
}

// sparsemax(z = Y * prior) via bisection on tau (sum(max(z-tau,0)) = 1 is
// monotone, root in [max-1, max]); then prior *= (GAMMA - mask) and
// Aout = bf16(x * mask) for the next shared-GLU GEMM.
__device__ __forceinline__ void sparsemask(float (*Y)[256], float (*prior)[256],
                                           const unsigned short (*xs)[256],
                                           unsigned short (*Aout)[256], int lane)
{
    for (int r = 0; r < ROWS; ++r) {
        float mx = -1e30f;
#pragma unroll
        for (int i = 0; i < 8; ++i) {
            int j = lane + 32 * i;
            float z = Y[r][j] * prior[r][j];
            Y[r][j] = z;
            mx = fmaxf(mx, z);
        }
#pragma unroll
        for (int o = 16; o; o >>= 1) mx = fmaxf(mx, __shfl_xor(mx, o));
#pragma unroll
        for (int i = 0; i < 8; ++i) Y[r][lane + 32 * i] -= mx;

        float lo = -1.f, hi = 0.f;
        for (int it = 0; it < 20; ++it) {          // tau to ~1e-6
            float tau = 0.5f * (lo + hi), s = 0.f;
#pragma unroll
            for (int i = 0; i < 8; ++i)
                s += fmaxf(Y[r][lane + 32 * i] - tau, 0.f);
#pragma unroll
            for (int o = 16; o; o >>= 1) s += __shfl_xor(s, o);
            if (s > 1.f) lo = tau; else hi = tau;
        }
        float tau = 0.5f * (lo + hi);
#pragma unroll
        for (int i = 0; i < 8; ++i) {
            int j = lane + 32 * i;
            float mask = fmaxf(Y[r][j] - tau, 0.f);
            prior[r][j] *= (TB_GAMMA - mask);
            Aout[r][j] = f2bf(bf2f(xs[r][j]) * mask);
        }
    }
}

// ------------------------------ fused kernel -------------------------------
__global__ __launch_bounds__(32) void tabnet_fused(
    const float* __restrict__ x,
    const unsigned short* __restrict__ wbf,   // converted bf16 weights
    const float* __restrict__ sh_b, const float* __restrict__ sh_lw,
    const float* __restrict__ sh_lb,
    const float* __restrict__ ft_b, const float* __restrict__ ft_lw,
    const float* __restrict__ ft_lb,
    const float* __restrict__ at_b, const float* __restrict__ at_lw,
    const float* __restrict__ at_lb,
    const float* __restrict__ fc_w, const float* __restrict__ fc_b,
    float* __restrict__ out)
{
    extern __shared__ __align__(16) char smem[];
    unsigned short (*xs)[256]   = (unsigned short (*)[256])(smem + L_XS);
    unsigned short (*abuf)[256] = (unsigned short (*)[256])(smem + L_ABUF);
    float (*ybuf)[256]  = (float (*)[256])(smem + L_YBUF);
    float (*prior)[256] = (float (*)[256])(smem + L_PRIOR);
    float (*oagg)[128]  = (float (*)[128])(smem + L_OAGG);

    const int lane = threadIdx.x;
    const int row0 = blockIdx.x * ROWS;

    const unsigned short* shT = wbf + OFF_SH;       // [256][256]
    const unsigned short* ftT = wbf + OFF_FT;       // [4][256][128]
    const unsigned short* atT = wbf + OFF_AT;       // [3][256][128]

    // Load x tile -> bf16, init prior / out_agg / A operand.
    for (int r = 0; r < ROWS; ++r) {
#pragma unroll
        for (int i = 0; i < 8; ++i) {
            int j = lane + 32 * i;
            float v = x[(size_t)(row0 + r) * TB_D + j];
            unsigned short h = f2bf(v);
            xs[r][j] = h;
            abuf[r][j] = h;
            prior[r][j] = 1.f;
        }
#pragma unroll
        for (int i = 0; i < 4; ++i) oagg[r][lane + 32 * i] = 0.f;
    }
    __syncthreads();

    // ---- initial transform: feat_tf(x, 0) (no accumulation) ----
    wmma_gemm(abuf, shT, 256, 256, sh_b, ybuf, lane);
    ln_glu(ybuf, sh_lw, sh_lb, abuf, oagg, lane, false);
    wmma_gemm(abuf, ftT + 0 * 256 * 128, 128, 256, ft_b + 0 * 256, ybuf, lane);
    ln_glu(ybuf, ft_lw + 0 * 256, ft_lb + 0 * 256, abuf, oagg, lane, false);

    // ---- decision steps ----
    for (int s = 0; s < TB_S; ++s) {
        wmma_gemm(abuf, atT + (size_t)s * 256 * 128, 128, 256,
                  at_b + s * 256, ybuf, lane);
        ln_inplace(ybuf, at_lw + s * 256, at_lb + s * 256, lane);
        sparsemask(ybuf, prior, xs, abuf, lane);
        wmma_gemm(abuf, shT, 256, 256, sh_b, ybuf, lane);
        ln_glu(ybuf, sh_lw, sh_lb, abuf, oagg, lane, false);
        wmma_gemm(abuf, ftT + (size_t)(s + 1) * 256 * 128, 128, 256,
                  ft_b + (s + 1) * 256, ybuf, lane);
        ln_glu(ybuf, ft_lw + (s + 1) * 256, ft_lb + (s + 1) * 256,
               abuf, oagg, lane, true);   // out_agg += x_t
    }
    __syncthreads();

    // ---- classifier: logits = out_agg @ fc_w + fc_b (tiny -> VALU) ----
    for (int idx = lane; idx < ROWS * TB_C; idx += 32) {
        int r = idx / TB_C, c = idx % TB_C;
        float acc = fc_b[c];
#pragma unroll 4
        for (int k = 0; k < TB_H; ++k)
            acc += oagg[r][k] * fc_w[k * TB_C + c];
        out[(size_t)(row0 + r) * TB_C + c] = acc;
    }
}

// ------------------------------- launcher ----------------------------------
extern "C" void kernel_launch(void* const* d_in, const int* in_sizes, int n_in,
                              void* d_out, int out_size, void* d_ws, size_t ws_size,
                              hipStream_t stream) {
    const float* x     = (const float*)d_in[0];
    const float* sh_w  = (const float*)d_in[1];
    const float* sh_b  = (const float*)d_in[2];
    const float* sh_lw = (const float*)d_in[3];
    const float* sh_lb = (const float*)d_in[4];
    const float* ft_w  = (const float*)d_in[5];
    const float* ft_b  = (const float*)d_in[6];
    const float* ft_lw = (const float*)d_in[7];
    const float* ft_lb = (const float*)d_in[8];
    const float* at_w  = (const float*)d_in[9];
    const float* at_b  = (const float*)d_in[10];
    const float* at_lw = (const float*)d_in[11];
    const float* at_lb = (const float*)d_in[12];
    const float* fc_w  = (const float*)d_in[13];
    const float* fc_b  = (const float*)d_in[14];
    float* out = (float*)d_out;
    unsigned short* wbf = (unsigned short*)d_ws;   // needs TOT_W*2 = 576 KB

    (void)in_sizes; (void)n_in; (void)out_size; (void)ws_size;

    // Allow 112 KB dynamic LDS (CDNA5 WGP has 320 KB). Host-side, idempotent,
    // not a stream op -> graph-capture safe.
    (void)hipFuncSetAttribute((const void*)tabnet_fused,
                              hipFuncAttributeMaxDynamicSharedMemorySize,
                              L_TOTAL);

    // 1) one-shot weight convert (re-done every call: deterministic)
    tabnet_convert_wts<<<(TOT_W + 255) / 256, 256, 0, stream>>>(sh_w, ft_w, at_w, wbf);

    // 2) fused forward: one wave32 per 32-row tile
    tabnet_fused<<<TB_B / ROWS, 32, L_TOTAL, stream>>>(
        x, wbf, sh_b, sh_lw, sh_lb, ft_b, ft_lw, ft_lb,
        at_b, at_lw, at_lb, fc_w, fc_b, out);
}
